// HeteroIsoNode_53910429499426
// MI455X (gfx1250) — matrix operand order repaired and checked by reading, not compiled, not run on hardware
//
#include <hip/hip_runtime.h>

#ifndef __has_builtin
#define __has_builtin(x) 0
#endif

typedef __attribute__((ext_vector_type(16))) __bf16   v16bf;
typedef __attribute__((ext_vector_type(8)))  float    v8f;
typedef __attribute__((ext_vector_type(4)))  unsigned u32x4;
typedef __attribute__((ext_vector_type(8)))  unsigned u32x8;

#define B_   1024
#define N_   160
#define IN_  256
#define OUT_ 256
#define NT_  8

// ---------------- CDNA5 helpers ----------------
__device__ __forceinline__ void async_ld_b128(unsigned lds_off, const void* gaddr) {
  // GLOBAL_LOAD_ASYNC_TO_LDS_B128: vdst = LDS byte address, vaddr = 64-bit global
  asm volatile("global_load_async_to_lds_b128 %0, %1, off"
               :: "v"(lds_off), "v"(gaddr) : "memory");
}
__device__ __forceinline__ void wait_async0() {
#if __has_builtin(__builtin_amdgcn_s_wait_asynccnt)
  __builtin_amdgcn_s_wait_asynccnt(0);
#else
  asm volatile("s_wait_asynccnt 0" ::: "memory");
#endif
}
__device__ __forceinline__ void wait_tensor0() {
#if __has_builtin(__builtin_amdgcn_s_wait_tensorcnt)
  __builtin_amdgcn_s_wait_tensorcnt(0);
#else
  asm volatile("s_wait_tensorcnt 0" ::: "memory");
#endif
}
__device__ __forceinline__ v16bf pack16(float4 a, float4 b, float4 c, float4 d) {
  v16bf r;
  r[0]=(__bf16)a.x;  r[1]=(__bf16)a.y;  r[2]=(__bf16)a.z;  r[3]=(__bf16)a.w;
  r[4]=(__bf16)b.x;  r[5]=(__bf16)b.y;  r[6]=(__bf16)b.z;  r[7]=(__bf16)b.w;
  r[8]=(__bf16)c.x;  r[9]=(__bf16)c.y;  r[10]=(__bf16)c.z; r[11]=(__bf16)c.w;
  r[12]=(__bf16)d.x; r[13]=(__bf16)d.y; r[14]=(__bf16)d.z; r[15]=(__bf16)d.w;
  return r;
}

// ---------------- Kernel 0: fold gcn_W into the 8 typed weights ----------------
// Wc[t] = f_W[t] @ gcn_W   (stored bf16, pre-swizzled into WMMA B-fragment order)
// bc[t] = f_b[t] @ gcn_W + gcn_b  (fp32)
__global__ __launch_bounds__(256) void prep_kernel(
    const float* __restrict__ fW, const float* __restrict__ fb,
    const float* __restrict__ gW, const float* __restrict__ gb,
    __bf16* __restrict__ wcf, float* __restrict__ bc) {
  int gid = blockIdx.x * 256 + threadIdx.x;           // 8*256*256 threads
  int t = gid >> 16, i = (gid >> 8) & 255, o = gid & 255;
  const float* wrow = fW + (size_t)(t * 256 + i) * 256;
  float acc = 0.f;
  for (int m = 0; m < 256; ++m) acc = fmaf(wrow[m], gW[m * 256 + o], acc);
  // B-fragment swizzle: K-row i -> (ktile, lane-half, element), N-col o -> (ntile, lane)
  int kt = i >> 5, kin = i & 31, nt = o >> 4;
  int lane = (o & 15) + ((kin >= 16) ? 16 : 0);
  int e = kin & 15;
  wcf[(size_t)(((t * 8 + kt) * 16 + nt) * 32 + lane) * 16 + e] = (__bf16)acc;
  if (gid < NT_ * OUT_) {
    int tt = gid >> 8, oo = gid & 255;
    float a2 = gb[oo];
    for (int m = 0; m < 256; ++m) a2 = fmaf(fb[tt * 256 + m], gW[m * 256 + oo], a2);
    bc[gid] = a2;
  }
}

// ---------------- Kernel 1: h2 = feat @ Wc[type] + bc[type]  (bf16 out) ----------------
// One block per batch. 8 waves, each owns 2 N-tiles. feat tiles staged via async->LDS.
__global__ __launch_bounds__(256) void typed_gemm_kernel(
    const float* __restrict__ feat, const int* __restrict__ types,
    const __bf16* __restrict__ wcf, const float* __restrict__ bc,
    __bf16* __restrict__ h2f) {
  __shared__ float sfeat[16 * 256];    // 16 KB row tile
  __shared__ int stypes[16];
  const int b = blockIdx.x;
  const int tid = threadIdx.x, lane = tid & 31, w = tid >> 5;
  const int hi = lane >> 4, rlo = lane & 15;
  const unsigned lds_base = (unsigned)(size_t)(&sfeat[0]);
  const v16bf* wv = (const v16bf*)wcf;

  for (int rt = 0; rt < 10; ++rt) {
    const float* src = feat + (size_t)(b * N_ + rt * 16) * IN_;
    for (int c = tid; c < 1024; c += 256)                 // 16x256 f32 = 1024 x b128
      async_ld_b128(lds_base + (unsigned)c * 16u, src + c * 4);
    if (tid < 16) stypes[tid] = types[b * N_ + rt * 16 + tid];
    wait_async0();
    __syncthreads();

    int ty[16]; unsigned tmask = 0;
#pragma unroll
    for (int i = 0; i < 16; ++i) { ty[i] = stypes[i]; tmask |= 1u << (ty[i] & 7); }

    // A fragments for all 8 K-tiles (contiguous A-layout: lane<16 -> K {0..15})
    v16bf afr[8];
#pragma unroll
    for (int kt = 0; kt < 8; ++kt) {
      const float* p = &sfeat[rlo * 256 + kt * 32 + hi * 16];
      afr[kt] = pack16(*(const float4*)(p),      *(const float4*)(p + 4),
                       *(const float4*)(p + 8),  *(const float4*)(p + 12));
    }

    for (int nn = 0; nn < 2; ++nn) {
      const int ntile = w * 2 + nn;
      float acc[8] = {0, 0, 0, 0, 0, 0, 0, 0};
      for (int ti = 0; ti < NT_; ++ti) {
        if (!((tmask >> ti) & 1u)) continue;              // skip absent types
        v8f c = {0.f, 0.f, 0.f, 0.f, 0.f, 0.f, 0.f, 0.f};
#pragma unroll
        for (int kt = 0; kt < 8; ++kt) {
          v16bf bfr = wv[(size_t)(((ti * 8 + kt) * 16 + ntile) * 32 + lane)];
          c = __builtin_amdgcn_wmma_f32_16x16x32_bf16(false, afr[kt], false, bfr,
                                                      (short)0, c, false, false);
        }
        const float bias = bc[ti * OUT_ + ntile * 16 + rlo];
#pragma unroll
        for (int j = 0; j < 8; ++j) {                     // C-layout: row j / j+8
          int r = j + hi * 8;
          if (ty[r] == ti) acc[j] = c[j] + bias;
        }
      }
      // store bf16 h2 pre-swizzled into stage-3 B-fragment order
#pragma unroll
      for (int j = 0; j < 8; ++j) {
        int m = rt * 16 + j + hi * 8;                     // node index (K dim of stage 3)
        int o = ntile * 16 + rlo;                         // feature col (N dim of stage 3)
        int kt2 = m >> 5, hb = (m >> 4) & 1, e = m & 15;
        size_t idx = ((size_t)(((b * 5 + kt2) * 16 + (o >> 4)) * 32 + (o & 15) + hb * 16)) * 16 + e;
        h2f[idx] = (__bf16)acc[j];
      }
    }
    __syncthreads();
  }
}

// ---------------- Kernel 2: out = relu(g @ h2)  per batch ----------------
// Whole 80 KB h2[b] fragment tile DMA'd into LDS by the Tensor Data Mover.
__global__ __launch_bounds__(256) void gcn_agg_kernel(
    const float* __restrict__ g, const __bf16* __restrict__ h2f,
    float* __restrict__ out) {
  __shared__ v16bf sh2[5 * 16 * 32];                     // 80 KB (320 KB WGP LDS)
  const int b = blockIdx.x;
  const int tid = threadIdx.x, lane = tid & 31, w = tid >> 5;
  const int hi = lane >> 4, rlo = lane & 15;

  if (tid < 32) {                                        // wave 0 issues one TDM op
    const unsigned lds_addr = (unsigned)(size_t)(&sh2[0]);
    const unsigned long long ga =
        (unsigned long long)(size_t)(h2f + (size_t)b * (5 * 16 * 32 * 16));
    const unsigned elems = (5 * 16 * 32 * 16 * 2) / 8;   // 81920 B in 8B units = 10240
    u32x4 d0; u32x8 d1;
    d0[0] = 1u;                                          // count=1, user descriptor
    d0[1] = lds_addr;                                    // lds_addr[31:0]
    d0[2] = (unsigned)ga;                                // global_addr[31:0]
    d0[3] = (unsigned)((ga >> 32) & 0x01FFFFFFu) | (2u << 30);  // addr[56:32] | type=2
    d1[0] = 3u << 16;                                    // data_size = 8 bytes
    d1[1] = (elems & 0xFFFFu) << 16;                     // tensor_dim0[15:0]
    d1[2] = (elems >> 16) | (1u << 16);                  // tensor_dim0[31:16] | tensor_dim1=1
    d1[3] = (elems & 0xFFFFu) << 16;                     // tile_dim0 = 10240
    d1[4] = 1u;                                          // tile_dim1 = 1
    d1[5] = elems;                                       // tensor_dim0_stride
    d1[6] = (elems & 0xFFFFu) << 16;                     // tensor_dim1_stride lo
    d1[7] = (elems >> 16);                               // tensor_dim1_stride hi
    asm volatile("tensor_load_to_lds %0, %1" :: "s"(d0), "s"(d1) : "memory");
    wait_tensor0();
  }
  __syncthreads();

  const int nmt = (w < 2) ? 2 : 1;                       // 10 row-tiles over 8 waves
  for (int im = 0; im < nmt; ++im) {
    const int mt = w + im * 8;
    v16bf afr[5];
    const float* gr = g + (size_t)(b * N_ + mt * 16 + rlo) * N_;
#pragma unroll
    for (int kt = 0; kt < 5; ++kt) {                     // K = 160 = 5 x 32
      const float* p = gr + kt * 32 + hi * 16;
      afr[kt] = pack16(*(const float4*)(p),      *(const float4*)(p + 4),
                       *(const float4*)(p + 8),  *(const float4*)(p + 12));
    }
    for (int ntile = 0; ntile < 16; ++ntile) {
      v8f c = {0.f, 0.f, 0.f, 0.f, 0.f, 0.f, 0.f, 0.f};
#pragma unroll
      for (int kt = 0; kt < 5; ++kt) {
        v16bf bfr = sh2[(kt * 16 + ntile) * 32 + lane];  // one ds_load_b128 pair
        c = __builtin_amdgcn_wmma_f32_16x16x32_bf16(false, afr[kt], false, bfr,
                                                    (short)0, c, false, false);
      }
      const int o = ntile * 16 + rlo;
      float* op = out + (size_t)(b * N_ + mt * 16 + hi * 8) * OUT_ + o;
#pragma unroll
      for (int j = 0; j < 8; ++j) {
        float v = c[j];
        op[(size_t)j * OUT_] = v > 0.f ? v : 0.f;        // ReLU
      }
    }
  }
}

// ---------------- Host launcher ----------------
extern "C" void kernel_launch(void* const* d_in, const int* in_sizes, int n_in,
                              void* d_out, int out_size, void* d_ws, size_t ws_size,
                              hipStream_t stream) {
  (void)in_sizes; (void)n_in; (void)out_size; (void)ws_size;
  const float* g     = (const float*)d_in[0];
  const int*   types = (const int*)  d_in[1];
  const float* feat  = (const float*)d_in[2];
  const float* fW    = (const float*)d_in[3];
  const float* fb    = (const float*)d_in[4];
  const float* gW    = (const float*)d_in[5];
  const float* gb    = (const float*)d_in[6];
  float* out = (float*)d_out;
  char*  ws  = (char*)d_ws;
  __bf16* wcf = (__bf16*)(ws);                 // 1 MB   folded weights (bf16, swizzled)
  float*  bc  = (float*) (ws + (1u << 20));    // 8 KB   folded biases
  __bf16* h2f = (__bf16*)(ws + (2u << 20));    // 84 MB  h2 intermediate (bf16, swizzled)

  prep_kernel<<<(NT_ * IN_ * OUT_) / 256, 256, 0, stream>>>(fW, fb, gW, gb, wcf, bc);
  typed_gemm_kernel<<<B_, 256, 0, stream>>>(feat, types, wcf, bc, h2f);
  gcn_agg_kernel<<<B_, 256, 0, stream>>>(g, h2f, out);
}
